// GQAModel_19963007991893
// MI455X (gfx1250) — matrix operand order repaired and checked by reading, not compiled, not run on hardware
//
#include <hip/hip_runtime.h>

typedef __attribute__((ext_vector_type(16))) _Float16 v16h;
typedef __attribute__((ext_vector_type(8)))  _Float16 v8h;
typedef __attribute__((ext_vector_type(8)))  float    v8f;
typedef __attribute__((ext_vector_type(4)))  float    f32x4;

namespace {
constexpr int Bb  = 2;
constexpr int Hh  = 32;
constexpr int Hkv = 8;
constexpr int Ss  = 2048;
constexpr int Dd  = 128;
constexpr int Gg  = Hh / Hkv;     // 4 query heads per KV head
constexpr int BM  = 128;          // Q rows per block (8 waves x 16)
constexpr int BN  = 64;           // keys per tile
constexpr int NW  = 8;            // waves per block
constexpr int KSTR = Dd + 8;      // 136 halfs, avoids LDS bank conflicts
constexpr int VSTR = BN + 8;      // 72
constexpr int PSTR = BN + 8;      // 72
constexpr float SCALE = 0.08838834764831845f; // 1/sqrt(128)
}

// 16 contiguous halfs -> v16h (two 16B LDS loads, 16B aligned)
__device__ __forceinline__ v16h ld16c(const _Float16* p) {
  v8h a = *(const v8h*)p;
  v8h b = *(const v8h*)(p + 8);
  v16h r;
#pragma unroll
  for (int i = 0; i < 8; ++i) { r[i] = a[i]; r[i + 8] = b[i]; }
  return r;
}

// A-layout gap load: halfs [0..7] and [16..23]
__device__ __forceinline__ v16h ld16g(const _Float16* p) {
  v8h a = *(const v8h*)p;
  v8h b = *(const v8h*)(p + 16);
  v16h r;
#pragma unroll
  for (int i = 0; i < 8; ++i) { r[i] = a[i]; r[i + 8] = b[i]; }
  return r;
}

__global__ __launch_bounds__(256, 1)
void gqa_flash_fwd(const float* __restrict__ q, const float* __restrict__ k,
                   const float* __restrict__ v, float* __restrict__ out)
{
  __shared__ _Float16 kl[BN * KSTR];      // K tile, row-major [key][d]
  __shared__ _Float16 vt[Dd * VSTR];      // V tile, transposed [d][key]
  __shared__ _Float16 pl[NW * 16 * PSTR]; // per-wave P scratch [row][key]

  const int tid  = threadIdx.x;
  const int wave = tid >> 5;
  const int lane = tid & 31;
  const int n16  = lane & 15;
  const int hsel = lane >> 4;

  constexpr int QT = Ss / BM;  // 16 q-tiles per (b,h)
  const int bid = blockIdx.x;
  const int qt  = bid % QT;
  const int bh  = bid / QT;
  const int h   = bh % Hh;
  const int b   = bh / Hh;
  const int hk  = h / Gg;

  const float* qp = q + ((size_t)(b * Hh  + h )) * Ss * Dd;
  const float* kp = k + ((size_t)(b * Hkv + hk)) * Ss * Dd;
  const float* vp = v + ((size_t)(b * Hkv + hk)) * Ss * Dd;
  float*       op = out + ((size_t)(b * Hh + h)) * Ss * Dd;

  const int qrow0 = qt * BM + wave * 16;

  // ---- Q A-fragments (16x32 f16), pre-scaled by 1/sqrt(D) ----
  // lane: row m = lane&15; element i<8 -> K = hsel*8+i ; i>=8 -> K = 16+hsel*8+(i-8)
  v16h qa[4];
#pragma unroll
  for (int kc = 0; kc < 4; ++kc) {
    const float* qr = qp + (size_t)(qrow0 + n16) * Dd + kc * 32 + hsel * 8;
    f32x4 a0 = *(const f32x4*)(qr + 0);
    f32x4 a1 = *(const f32x4*)(qr + 4);
    f32x4 b0 = *(const f32x4*)(qr + 16);
    f32x4 b1 = *(const f32x4*)(qr + 20);
    v16h f;
#pragma unroll
    for (int i = 0; i < 4; ++i) {
      f[i]      = (_Float16)(a0[i] * SCALE);
      f[4 + i]  = (_Float16)(a1[i] * SCALE);
      f[8 + i]  = (_Float16)(b0[i] * SCALE);
      f[12 + i] = (_Float16)(b1[i] * SCALE);
    }
    qa[kc] = f;
  }

  // ---- accumulators: O (16x128 f32 as 8 C-fragments), row stats ----
  v8f o[8];
#pragma unroll
  for (int dt = 0; dt < 8; ++dt) o[dt] = {};
  float m_i[8], l_i[8];
#pragma unroll
  for (int j = 0; j < 8; ++j) { m_i[j] = -3.0e38f; l_i[j] = 0.0f; }

  const int nkt = (qt + 1) * (BM / BN);  // causal clip at block granularity

  for (int kt = 0; kt < nkt; ++kt) {
    const int kbase = kt * BN;

    __syncthreads();  // previous tile fully consumed
    {
      // cooperative K/V tile load: 4 threads per key row, 32 d each
      const int r  = tid >> 2;
      const int d0 = (tid & 3) * 32;
      const float* kr = kp + (size_t)(kbase + r) * Dd + d0;
      const float* vr = vp + (size_t)(kbase + r) * Dd + d0;
#pragma unroll
      for (int i = 0; i < 32; i += 4) {
        f32x4 kk = *(const f32x4*)(kr + i);
        f32x4 vv = *(const f32x4*)(vr + i);
#pragma unroll
        for (int j = 0; j < 4; ++j) {
          kl[r * KSTR + d0 + i + j]   = (_Float16)kk[j];
          vt[(d0 + i + j) * VSTR + r] = (_Float16)vv[j];
        }
      }
      if (kt + 1 < nkt) {  // warm next tile into near caches (WGP scope)
        __builtin_prefetch(kr + (size_t)BN * Dd, 0, 3);
        __builtin_prefetch(vr + (size_t)BN * Dd, 0, 3);
      }
    }
    __syncthreads();

    if (kbase > qrow0 + 15) continue;  // tile fully above causal diagonal for this wave

    // ---- scores S = (Q*scale) K^T : 4 col-tiles x 4 K-chunks of 32 ----
    // Preload all 4 B-fragments of a tile into distinct regs so the ds_loads
    // of tile c+1 can issue underneath the WMMAs of tile c (partial dscnt waits).
    v8f sc[4];
#pragma unroll
    for (int c = 0; c < 4; ++c) {
      const _Float16* kb = &kl[(c * 16 + n16) * KSTR + hsel * 8];
      v16h kb0 = ld16g(kb + 0 * 32);
      v16h kb1 = ld16g(kb + 1 * 32);
      v16h kb2 = ld16g(kb + 2 * 32);
      v16h kb3 = ld16g(kb + 3 * 32);
      v8f acc = {};
      acc = __builtin_amdgcn_wmma_f32_16x16x32_f16(false, qa[0], false, kb0,
                                                   (short)0, acc, false, false);
      acc = __builtin_amdgcn_wmma_f32_16x16x32_f16(false, qa[1], false, kb1,
                                                   (short)0, acc, false, false);
      acc = __builtin_amdgcn_wmma_f32_16x16x32_f16(false, qa[2], false, kb2,
                                                   (short)0, acc, false, false);
      acc = __builtin_amdgcn_wmma_f32_16x16x32_f16(false, qa[3], false, kb3,
                                                   (short)0, acc, false, false);
      sc[c] = acc;
    }

    // ---- causal mask + online softmax (C layout: row = j + 8*hsel, col = n16) ----
    float mnew[8];
#pragma unroll
    for (int j = 0; j < 8; ++j) mnew[j] = m_i[j];
#pragma unroll
    for (int c = 0; c < 4; ++c) {
      const int col = kbase + c * 16 + n16;
#pragma unroll
      for (int j = 0; j < 8; ++j) {
        const int row = qrow0 + j + 8 * hsel;
        float s = sc[c][j];
        s = (col > row) ? -3.0e38f : s;
        sc[c][j] = s;
        mnew[j]  = fmaxf(mnew[j], s);
      }
    }
#pragma unroll
    for (int j = 0; j < 8; ++j) {   // row-max across the 16 lanes of each half
      float mv = mnew[j];
#pragma unroll
      for (int off = 1; off < 16; off <<= 1)
        mv = fmaxf(mv, __shfl_xor(mv, off, 32));
      mnew[j] = mv;
    }
    float alpha[8];
#pragma unroll
    for (int j = 0; j < 8; ++j) {
      alpha[j] = __expf(m_i[j] - mnew[j]);
      m_i[j]   = mnew[j];
      l_i[j]  *= alpha[j];
    }
#pragma unroll
    for (int dt = 0; dt < 8; ++dt)
#pragma unroll
      for (int j = 0; j < 8; ++j)
        o[dt][j] *= alpha[j];

    // ---- p = exp(s - m): accumulate row sums, spill to per-wave LDS as f16 ----
    float rs[8];
#pragma unroll
    for (int j = 0; j < 8; ++j) rs[j] = 0.0f;
#pragma unroll
    for (int c = 0; c < 4; ++c) {
#pragma unroll
      for (int j = 0; j < 8; ++j) {
        float p = __expf(sc[c][j] - m_i[j]);
        rs[j] += p;
        pl[(wave * 16 + j + 8 * hsel) * PSTR + c * 16 + n16] = (_Float16)p;
      }
    }
#pragma unroll
    for (int j = 0; j < 8; ++j) {
      float rv = rs[j];
#pragma unroll
      for (int off = 1; off < 16; off <<= 1)
        rv += __shfl_xor(rv, off, 32);
      l_i[j] += rv;
    }

    // ---- O += P V : 2 K-chunks of 32 keys x 8 d-tiles ----
    // same-wave LDS ops are in-order: pl write->read needs no barrier.
    // Preload V B-fragments 4 at a time to break the wait0-per-WMMA chain.
#pragma unroll
    for (int kc2 = 0; kc2 < 2; ++kc2) {
      v16h pa = ld16g(&pl[(wave * 16 + n16) * PSTR + kc2 * 32 + hsel * 8]);
#pragma unroll
      for (int dg = 0; dg < 2; ++dg) {
        const int dt0 = dg * 4;
        v16h vb0 = ld16g(&vt[((dt0 + 0) * 16 + n16) * VSTR + kc2 * 32 + hsel * 8]);
        v16h vb1 = ld16g(&vt[((dt0 + 1) * 16 + n16) * VSTR + kc2 * 32 + hsel * 8]);
        v16h vb2 = ld16g(&vt[((dt0 + 2) * 16 + n16) * VSTR + kc2 * 32 + hsel * 8]);
        v16h vb3 = ld16g(&vt[((dt0 + 3) * 16 + n16) * VSTR + kc2 * 32 + hsel * 8]);
        o[dt0 + 0] = __builtin_amdgcn_wmma_f32_16x16x32_f16(false, pa, false, vb0,
                                                            (short)0, o[dt0 + 0], false, false);
        o[dt0 + 1] = __builtin_amdgcn_wmma_f32_16x16x32_f16(false, pa, false, vb1,
                                                            (short)0, o[dt0 + 1], false, false);
        o[dt0 + 2] = __builtin_amdgcn_wmma_f32_16x16x32_f16(false, pa, false, vb2,
                                                            (short)0, o[dt0 + 2], false, false);
        o[dt0 + 3] = __builtin_amdgcn_wmma_f32_16x16x32_f16(false, pa, false, vb3,
                                                            (short)0, o[dt0 + 3], false, false);
      }
    }
  }

  // ---- epilogue: normalize and store fp32 ----
#pragma unroll
  for (int j = 0; j < 8; ++j) {
    const float inv = 1.0f / l_i[j];
    const int   row = qrow0 + j + 8 * hsel;
#pragma unroll
    for (int dt = 0; dt < 8; ++dt)
      op[(size_t)row * Dd + dt * 16 + n16] = o[dt][j] * inv;
  }
}

extern "C" void kernel_launch(void* const* d_in, const int* in_sizes, int n_in,
                              void* d_out, int out_size, void* d_ws, size_t ws_size,
                              hipStream_t stream) {
  (void)in_sizes; (void)n_in; (void)out_size; (void)d_ws; (void)ws_size;
  const float* q = (const float*)d_in[0];
  const float* k = (const float*)d_in[1];
  const float* v = (const float*)d_in[2];
  float* out = (float*)d_out;
  dim3 grid(Bb * Hh * (Ss / BM));  // 1024 workgroups
  dim3 block(256);                 // 8 waves (wave32)
  hipLaunchKernelGGL(gqa_flash_fwd, grid, block, 0, stream, q, k, v, out);
}
